// LinearCrossEntropyLoss_72902774882612
// MI455X (gfx1250) — compile-verified, matrix operand order checked by reading, not run on hardware
//
#include <hip/hip_runtime.h>
#include <hip/hip_bf16.h>
#include <math.h>

#define IGNORE_INDEX (-100)

typedef __attribute__((ext_vector_type(16))) __bf16 v16bf;
typedef __attribute__((ext_vector_type(8)))  __bf16 v8bf;
typedef __attribute__((ext_vector_type(4)))  __bf16 v4bf;
typedef __attribute__((ext_vector_type(8)))  float  v8f;

static constexpr int M_TOK = 8192;    // 4 * 2048 tokens
static constexpr int D_DIM = 2048;    // hidden
static constexpr int V_DIM = 32000;   // vocab
static constexpr int BM = 128;
static constexpr int BN = 128;
static constexpr int BK = 64;
static constexpr int LDK = BK + 8;    // padded LDS row stride: 144B -> 16 rows hit 16 distinct banks
static constexpr int TILE = BM * LDK;       // elems per plane tile
static constexpr int BUFE = 4 * TILE;       // elems per buffer (Ah, Al, Bh, Bl)
static constexpr int NSLICES = V_DIM / 64;  // one partial per 64-col wave slice = 500

// ---------------------------------------------------------------------------
// Kernel 0: one-shot f32 -> (bf16 hi, bf16 lo) split. Bandwidth-bound, tiny
// vs GEMM. Afterwards the GEMM streams pure bf16 (Whi fits in 192MB L2).
// ---------------------------------------------------------------------------
__global__ __launch_bounds__(256)
void split_bf16(const float4* __restrict__ src, v4bf* __restrict__ hi,
                v4bf* __restrict__ lo, int n4) {
    for (int i = blockIdx.x * 256 + threadIdx.x; i < n4; i += gridDim.x * 256) {
        float4 v = src[i];
        v4bf h, l;
        h[0] = (__bf16)v.x; l[0] = (__bf16)(v.x - (float)h[0]);
        h[1] = (__bf16)v.y; l[1] = (__bf16)(v.y - (float)h[1]);
        h[2] = (__bf16)v.z; l[2] = (__bf16)(v.z - (float)h[2]);
        h[3] = (__bf16)v.w; l[3] = (__bf16)(v.w - (float)h[3]);
        hi[i] = h;
        lo[i] = l;
    }
}

// ---------------------------------------------------------------------------
// WMMA fragment load from a row-major bf16 LDS tile (ISA 7.12.2 layout):
// lanes 0-15: row r, K[k0..k0+8) and K[k0+16..k0+24);
// lanes 16-31: row r, K[k0+8..k0+16) and K[k0+24..k0+32).
// ---------------------------------------------------------------------------
__device__ __forceinline__ v16bf load_frag(const __bf16* lds, int row, int k0, int lane) {
    const __bf16* base = lds + row * LDK + k0 + ((lane & 16) ? 8 : 0);
    v8bf a0 = *(const v8bf*)(base);        // ds_load_b128
    v8bf a1 = *(const v8bf*)(base + 16);   // ds_load_b128
    return __builtin_shufflevector(a0, a1, 0,1,2,3,4,5,6,7,8,9,10,11,12,13,14,15);
}

// Async-copy one 128x64 bf16 plane (global row-major, ld=D_DIM) into LDS.
// 1024 16B chunks, 4 per thread, via GLOBAL_LOAD_ASYNC_TO_LDS_B128 (ASYNCcnt).
__device__ __forceinline__ void stage_plane_async(const __bf16* __restrict__ gbase,
                                                  int row0, int kt,
                                                  uint32_t lds_byte_base, int tid) {
    #pragma unroll
    for (int j = 0; j < 4; ++j) {
        int c   = tid + j * 256;
        int row = c >> 3;            // 8 chunks per 64-col row
        int col = (c & 7) * 8;
        uint32_t goff = (uint32_t)(((size_t)(row0 + row) * D_DIM + kt + col) * sizeof(__bf16));
        uint32_t loff = lds_byte_base + (uint32_t)((row * LDK + col) * sizeof(__bf16));
        asm volatile("global_load_async_to_lds_b128 %0, %1, %2 offset:0"
                     :
                     : "v"(loff), "v"(goff), "s"(gbase)
                     : "memory");
    }
}

__device__ __forceinline__ void wait_async_all() {
    asm volatile("s_wait_asynccnt 0x0" ::: "memory");
}

// ---------------------------------------------------------------------------
// Kernel 1: double-buffered split-bf16 WMMA GEMM fused with per-(token,
// 64-col-slice) online-softmax partials (rowmax, sumexp) written to ws.
// ---------------------------------------------------------------------------
__global__ __launch_bounds__(256)
void lce_gemm_lse_partial(const __bf16* __restrict__ Whi, const __bf16* __restrict__ Wlo,
                          const __bf16* __restrict__ Xhi, const __bf16* __restrict__ Xlo,
                          float2* __restrict__ partial) {
    extern __shared__ __bf16 smem[];   // 2 * BUFE elems = 147456 B

    const int tid  = threadIdx.x;
    const int lane = tid & 31;
    const int wave = tid >> 5;     // 8 waves
    const int mw   = wave >> 1;    // 0..3  -> row offset mw*32
    const int nw   = wave & 1;     // 0..1  -> col offset nw*64
    const int m_base = blockIdx.x * BM;
    const int v_base = blockIdx.y * BN;

    const uint32_t sbase = (uint32_t)(uintptr_t)&smem[0];  // flat low 32 = LDS offset

    v8f acc[2][4];
    #pragma unroll
    for (int i = 0; i < 2; ++i)
        #pragma unroll
        for (int j = 0; j < 4; ++j)
            acc[i][j] = (v8f)0.0f;

    auto stage_buf = [&](int b, int kt) {
        uint32_t bb = sbase + (uint32_t)(b * BUFE * sizeof(__bf16));
        stage_plane_async(Xhi, m_base, kt, bb + 0 * TILE * sizeof(__bf16), tid);
        stage_plane_async(Xlo, m_base, kt, bb + 1 * TILE * sizeof(__bf16), tid);
        stage_plane_async(Whi, v_base, kt, bb + 2 * TILE * sizeof(__bf16), tid);
        stage_plane_async(Wlo, v_base, kt, bb + 3 * TILE * sizeof(__bf16), tid);
    };

    // prologue: fill buffer 0
    stage_buf(0, 0);
    wait_async_all();
    __syncthreads();

    int b = 0;
    for (int kt = 0; kt < D_DIM; kt += BK, b ^= 1) {
        // kick off next tile's copies into the other buffer (overlaps compute)
        if (kt + BK < D_DIM) stage_buf(b ^ 1, kt + BK);

        const __bf16* sAh = smem + b * BUFE + 0 * TILE;
        const __bf16* sAl = smem + b * BUFE + 1 * TILE;
        const __bf16* sBh = smem + b * BUFE + 2 * TILE;
        const __bf16* sBl = smem + b * BUFE + 3 * TILE;

        #pragma unroll
        for (int k0 = 0; k0 < BK; k0 += 32) {
            v16bf ah[2], al[2], bh[4], bl[4];
            #pragma unroll
            for (int mi = 0; mi < 2; ++mi) {
                int r = mw * 32 + mi * 16 + (lane & 15);
                ah[mi] = load_frag(sAh, r, k0, lane);
                al[mi] = load_frag(sAl, r, k0, lane);
            }
            #pragma unroll
            for (int ni = 0; ni < 4; ++ni) {
                int r = nw * 64 + ni * 16 + (lane & 15);
                bh[ni] = load_frag(sBh, r, k0, lane);
                bl[ni] = load_frag(sBl, r, k0, lane);
            }
            #pragma unroll
            for (int mi = 0; mi < 2; ++mi)
                #pragma unroll
                for (int ni = 0; ni < 4; ++ni) {
                    // fp32 ~= hi*hi + hi*lo + lo*hi  (lo*lo below fp32 ulp)
                    acc[mi][ni] = __builtin_amdgcn_wmma_f32_16x16x32_bf16(
                        false, ah[mi], false, bh[ni], (short)0, acc[mi][ni], false, false);
                    acc[mi][ni] = __builtin_amdgcn_wmma_f32_16x16x32_bf16(
                        false, ah[mi], false, bl[ni], (short)0, acc[mi][ni], false, false);
                    acc[mi][ni] = __builtin_amdgcn_wmma_f32_16x16x32_bf16(
                        false, al[mi], false, bh[ni], (short)0, acc[mi][ni], false, false);
                }
        }

        // next buffer's async copies done + all waves done reading this buffer
        wait_async_all();
        __syncthreads();
    }

    // ---- epilogue: per-row (max, sumexp) over this wave's 64 columns ----
    // C layout: VGPR j = row j (lanes 0-15) / row j+8 (lanes 16-31), col = lane%16.
    #pragma unroll
    for (int mi = 0; mi < 2; ++mi) {
        #pragma unroll
        for (int j = 0; j < 8; ++j) {
            float m = -__builtin_inff();
            #pragma unroll
            for (int ni = 0; ni < 4; ++ni) m = fmaxf(m, acc[mi][ni][j]);
            #pragma unroll
            for (int mask = 1; mask < 16; mask <<= 1)
                m = fmaxf(m, __shfl_xor(m, mask, 32));
            float s = 0.0f;
            #pragma unroll
            for (int ni = 0; ni < 4; ++ni) s += __expf(acc[mi][ni][j] - m);
            #pragma unroll
            for (int mask = 1; mask < 16; mask <<= 1)
                s += __shfl_xor(s, mask, 32);
            if ((lane & 15) == 0) {
                int row   = m_base + mw * 32 + mi * 16 + j + ((lane & 16) ? 8 : 0);
                int slice = blockIdx.y * 2 + nw;
                partial[(size_t)row * NSLICES + slice] = make_float2(m, s);
            }
        }
    }
}

// ---------------------------------------------------------------------------
// Kernel 2: target logit per token (one wave per token; reads original f32).
// ---------------------------------------------------------------------------
__global__ __launch_bounds__(256)
void lce_target_logit(const float* __restrict__ W, const float* __restrict__ X,
                      const int* __restrict__ T, float* __restrict__ tgt) {
    int wave  = threadIdx.x >> 5;
    int lane  = threadIdx.x & 31;
    int token = blockIdx.x * 8 + wave;
    int t  = T[token];
    int st = (t == IGNORE_INDEX) ? 0 : t;
    const float* x = X + (size_t)token * D_DIM;
    const float* w = W + (size_t)st * D_DIM;
    float acc = 0.0f;
    for (int k = lane * 4; k < D_DIM; k += 32 * 4) {
        float4 xv = *(const float4*)(x + k);
        float4 wv = *(const float4*)(w + k);
        acc += xv.x * wv.x + xv.y * wv.y + xv.z * wv.z + xv.w * wv.w;
    }
    #pragma unroll
    for (int mask = 1; mask < 32; mask <<= 1)
        acc += __shfl_xor(acc, mask, 32);
    if (lane == 0) tgt[token] = acc;
}

// ---------------------------------------------------------------------------
// Kernel 3: merge the 500 per-slice partials per token (online LSE) + mask.
// ---------------------------------------------------------------------------
__global__ __launch_bounds__(256)
void lce_combine(const float2* __restrict__ partial, const float* __restrict__ tgt,
                 const int* __restrict__ T, float* __restrict__ loss) {
    int token = blockIdx.x * 256 + threadIdx.x;
    if (token >= M_TOK) return;
    float M = -__builtin_inff(), S = 0.0f;
    const float2* p = partial + (size_t)token * NSLICES;
    for (int i = 0; i < NSLICES; ++i) {
        float2 v = p[i];
        float nm = fmaxf(M, v.x);
        S = S * __expf(M - nm) + v.y * __expf(v.x - nm);
        M = nm;
    }
    float lse = M + __logf(S);
    int t = T[token];
    loss[token] = (t == IGNORE_INDEX) ? 0.0f : (lse - tgt[token]);
}

// ---------------------------------------------------------------------------
// Kernel 4: masked mean -> scalar.
// ---------------------------------------------------------------------------
__global__ __launch_bounds__(256)
void lce_final(const float* __restrict__ loss, const int* __restrict__ T,
               float* __restrict__ out) {
    __shared__ float ssum[256];
    __shared__ int   scnt[256];
    float s = 0.0f; int c = 0;
    for (int i = threadIdx.x; i < M_TOK; i += 256) {
        s += loss[i];
        c += (T[i] != IGNORE_INDEX) ? 1 : 0;
    }
    ssum[threadIdx.x] = s; scnt[threadIdx.x] = c;
    __syncthreads();
    for (int step = 128; step > 0; step >>= 1) {
        if (threadIdx.x < step) {
            ssum[threadIdx.x] += ssum[threadIdx.x + step];
            scnt[threadIdx.x] += scnt[threadIdx.x + step];
        }
        __syncthreads();
    }
    if (threadIdx.x == 0) out[0] = ssum[0] / (float)scnt[0];
}

extern "C" void kernel_launch(void* const* d_in, const int* in_sizes, int n_in,
                              void* d_out, int out_size, void* d_ws, size_t ws_size,
                              hipStream_t stream) {
    (void)in_sizes; (void)n_in; (void)out_size; (void)ws_size;
    const float* W = (const float*)d_in[0];   // [32000, 2048] f32
    const float* X = (const float*)d_in[1];   // [4, 2048, 2048] f32 -> [8192, 2048]
    const int*   T = (const int*)d_in[2];     // [8192]
    float* out = (float*)d_out;

    const size_t WN = (size_t)V_DIM * D_DIM;  // 65,536,000 elems
    const size_t XN = (size_t)M_TOK * D_DIM;  // 16,777,216 elems

    char* ws = (char*)d_ws;
    float2* partial = (float2*)ws;                       // 8192*500*8   = 32.768 MB
    char*   p       = ws + (size_t)M_TOK * NSLICES * sizeof(float2);
    float*  tgt     = (float*)p;          p += (size_t)M_TOK * sizeof(float);
    float*  loss    = (float*)p;          p += (size_t)M_TOK * sizeof(float);
    __bf16* Whi     = (__bf16*)p;         p += WN * sizeof(__bf16);   // 131 MB
    __bf16* Wlo     = (__bf16*)p;         p += WN * sizeof(__bf16);   // 131 MB
    __bf16* Xhi     = (__bf16*)p;         p += XN * sizeof(__bf16);   //  33.5 MB
    __bf16* Xlo     = (__bf16*)p;

    // 0) split f32 -> bf16 hi/lo planes (bandwidth-bound)
    split_bf16<<<4096, 256, 0, stream>>>((const float4*)W, (v4bf*)Whi, (v4bf*)Wlo,
                                         (int)(WN / 4));
    split_bf16<<<4096, 256, 0, stream>>>((const float4*)X, (v4bf*)Xhi, (v4bf*)Xlo,
                                         (int)(XN / 4));

    // 1) fused WMMA GEMM + partial LSE (double-buffered async LDS staging)
    dim3 g1(M_TOK / BM, V_DIM / BN);   // 64 x 250
    size_t smem_bytes = (size_t)2 * BUFE * sizeof(__bf16);   // 147456 B
    lce_gemm_lse_partial<<<g1, 256, smem_bytes, stream>>>(Whi, Wlo, Xhi, Xlo, partial);

    // 2..4) target logits, per-token LSE merge, masked mean
    lce_target_logit<<<M_TOK / 8, 256, 0, stream>>>(W, X, T, tgt);
    lce_combine<<<(M_TOK + 255) / 256, 256, 0, stream>>>(partial, tgt, T, loss);
    lce_final<<<1, 256, 0, stream>>>(loss, T, out);
}
